// KNNAttentionSoftmaxOverLocalAndDistant_88390426952142
// MI455X (gfx1250) — matrix operand order repaired and checked by reading, not compiled, not run on hardware
//
#include <hip/hip_runtime.h>

// Problem constants (fixed by the reference)
#define B_  2
#define S_  1024
#define D_  1024
#define H_  16
#define DH_ 64
#define K_  32
#define M_  8192

typedef __attribute__((ext_vector_type(16))) __bf16 v16bf;
typedef __attribute__((ext_vector_type(8)))  __bf16 v8bf;
typedef __attribute__((ext_vector_type(8)))  float  v8f;

__device__ __forceinline__ v8f wmma_bf16(v16bf a, v16bf b, v8f c) {
  return __builtin_amdgcn_wmma_f32_16x16x32_bf16(false, a, false, b, (short)0, c, false, false);
}

__device__ __forceinline__ v16bf packA(v8bf lo, v8bf hi) {
  v16bf a;
#pragma unroll
  for (int i = 0; i < 8; ++i) { a[i] = lo[i]; a[i + 8] = hi[i]; }
  return a;
}

__device__ __forceinline__ float gelu_f(float x) {
  float x3 = x * x * x;
  float t = tanhf(0.7978845608028654f * (x + 0.044715f * x3));
  return 0.5f * x * (1.0f + t);
}

// ---------------------------------------------------------------------------
// Generic NT GEMM: C[m,n] = sum_k A[m,k] * Bt[n,k]  (+bias +addend, epilogue)
// A: (Mrows x Kdim) bf16 row-major; Bt: (N x Kdim) bf16 row-major.
// One wave per block; each wave computes a 16x64 tile (4 WMMA accumulators).
// EPI: 0 = none, 1 = tanh-gelu
// ---------------------------------------------------------------------------
template <int EPI>
__global__ __launch_bounds__(32)
void gemm_nt(const __bf16* __restrict__ A, const __bf16* __restrict__ Bt,
             const float* __restrict__ bias, const float* __restrict__ addend,
             float* __restrict__ Cf, __bf16* __restrict__ Cbf,
             int Mrows, int N, int Kdim,
             long sA, long sB, long sC) {
  const int lane = threadIdx.x;
  const int l15  = lane & 15;
  const int half = lane >> 4;
  const int n0   = blockIdx.x * 64;
  const int m0   = blockIdx.y * 16;
  const int b    = blockIdx.z;

  A  += (long)b * sA;
  Bt += (long)b * sB;
  const long cbase = (long)b * sC;

  const __bf16* arow = A + (long)(m0 + l15) * Kdim + half * 16;
  const __bf16* brow = Bt + (long)(n0 + l15) * Kdim + half * 16;
  const long bstep = (long)16 * Kdim;

  v8f acc0 = {}, acc1 = {}, acc2 = {}, acc3 = {};

  for (int k0 = 0; k0 < Kdim; k0 += 32) {
    __builtin_prefetch((const void*)(arow + k0 + 256), 0, 1);
    v8bf alo = *(const v8bf*)(arow + k0);
    v8bf ahi = *(const v8bf*)(arow + k0 + 8);
    v16bf a = packA(alo, ahi);
    v16bf b0 = *(const v16bf*)(brow + k0);
    v16bf b1 = *(const v16bf*)(brow + bstep + k0);
    v16bf b2 = *(const v16bf*)(brow + 2 * bstep + k0);
    v16bf b3 = *(const v16bf*)(brow + 3 * bstep + k0);
    acc0 = wmma_bf16(a, b0, acc0);
    acc1 = wmma_bf16(a, b1, acc1);
    acc2 = wmma_bf16(a, b2, acc2);
    acc3 = wmma_bf16(a, b3, acc3);
  }

  v8f accs[4] = {acc0, acc1, acc2, acc3};
  const int row_base = m0 + half * 8;
#pragma unroll
  for (int t = 0; t < 4; ++t) {
    const int col = n0 + t * 16 + l15;
    const float bcol = bias ? bias[col] : 0.0f;
#pragma unroll
    for (int r = 0; r < 8; ++r) {
      const long off = cbase + (long)(row_base + r) * N + col;
      float v = accs[t][r] + bcol;
      if (addend) v += addend[off];
      if (EPI == 1) v = gelu_f(v);
      if (Cf)  Cf[off]  = v;
      if (Cbf) Cbf[off] = (__bf16)v;
    }
  }
}

// ---------------------------------------------------------------------------
// LayerNorm: per row of D_ floats -> bf16 normalized output
// ---------------------------------------------------------------------------
__global__ __launch_bounds__(256)
void ln_kernel(const float* __restrict__ x, const float* __restrict__ g,
               const float* __restrict__ bt, __bf16* __restrict__ out) {
  const int Dn = D_;
  const long row = blockIdx.x;
  const float* xr = x + row * Dn;
  __shared__ float red[256];
  const int tid = threadIdx.x;

  float s = 0.0f;
  for (int i = tid; i < Dn; i += 256) s += xr[i];
  red[tid] = s; __syncthreads();
  for (int st = 128; st > 0; st >>= 1) { if (tid < st) red[tid] += red[tid + st]; __syncthreads(); }
  const float mean = red[0] / (float)Dn;
  __syncthreads();

  float v = 0.0f;
  for (int i = tid; i < Dn; i += 256) { float d = xr[i] - mean; v += d * d; }
  red[tid] = v; __syncthreads();
  for (int st = 128; st > 0; st >>= 1) { if (tid < st) red[tid] += red[tid + st]; __syncthreads(); }
  const float rstd = rsqrtf(red[0] / (float)Dn + 1e-5f);

  for (int i = tid; i < Dn; i += 256)
    out[row * Dn + i] = (__bf16)((xr[i] - mean) * rstd * g[i] + bt[i]);
}

// ---------------------------------------------------------------------------
// sq = q_f / max(||q_f||, 1e-12), bf16 out.  q_f = qkv[:, 0:D]
// ---------------------------------------------------------------------------
__global__ __launch_bounds__(256)
void rownorm_kernel(const float* __restrict__ qkv, __bf16* __restrict__ sq) {
  const int Dn = D_;
  const long row = blockIdx.x;
  const float* xr = qkv + row * (3 * D_);
  __shared__ float red[256];
  const int tid = threadIdx.x;
  float s = 0.0f;
  for (int i = tid; i < Dn; i += 256) { float q = xr[i]; s += q * q; }
  red[tid] = s; __syncthreads();
  for (int st = 128; st > 0; st >>= 1) { if (tid < st) red[tid] += red[tid + st]; __syncthreads(); }
  float nrm = sqrtf(red[0]);
  nrm = fmaxf(nrm, 1e-12f);
  const float inv = 1.0f / nrm;
  for (int i = tid; i < Dn; i += 256) sq[row * Dn + i] = (__bf16)(xr[i] * inv);
}

// ---------------------------------------------------------------------------
// Top-K (K=32) per row of sims (M_=8192) via iterative argmax over LDS.
// ---------------------------------------------------------------------------
__global__ __launch_bounds__(256)
void topk_kernel(const float* __restrict__ sims, int* __restrict__ idxout) {
  __shared__ unsigned int vals[M_];
  __shared__ unsigned long long red[256];
  const long row = blockIdx.x;
  const float* src = sims + row * M_;
  const int tid = threadIdx.x;

  for (int i = tid; i < M_; i += 256) {
    unsigned int u = __float_as_uint(src[i]);
    u = (u & 0x80000000u) ? ~u : (u | 0x80000000u); // monotone mapping
    vals[i] = u;
  }
  __syncthreads();

  for (int sel = 0; sel < K_; ++sel) {
    unsigned long long best = 0ull;
    for (int i = tid; i < M_; i += 256) {
      unsigned long long key = ((unsigned long long)vals[i] << 32) | (unsigned)i;
      if (key > best) best = key;
    }
    red[tid] = best; __syncthreads();
    for (int st = 128; st > 0; st >>= 1) {
      if (tid < st) red[tid] = (red[tid] > red[tid + st]) ? red[tid] : red[tid + st];
      __syncthreads();
    }
    if (tid == 0) {
      int ai = (int)(red[0] & 0xffffffffu);
      idxout[row * K_ + sel] = ai;
      vals[ai] = 0u; // remove from further consideration
    }
    __syncthreads();
  }
}

// ---------------------------------------------------------------------------
// Repack qkv fp32 -> per-head bf16: qh (scaled by 1/sqrt(DH)), kh, vt (DHxS)
// ---------------------------------------------------------------------------
__global__ __launch_bounds__(256)
void repack_kernel(const float* __restrict__ qkv, __bf16* __restrict__ qh,
                   __bf16* __restrict__ kh, __bf16* __restrict__ vt) {
  const long t = (long)blockIdx.x * 256 + threadIdx.x; // < B*H*S*64
  const int d = (int)(t & 63);
  const int s = (int)((t >> 6) & (S_ - 1));
  const int h = (int)((t >> 16) & (H_ - 1));
  const int b = (int)(t >> 20);
  const long qbase = (long)(b * S_ + s) * (3 * D_) + h * DH_ + d;
  qh[t] = (__bf16)(qkv[qbase] * 0.125f);      // 1/sqrt(64)
  kh[t] = (__bf16)(qkv[qbase + D_]);
  vt[((long)(b * H_ + h) * DH_ + d) * S_ + s] = (__bf16)(qkv[qbase + 2 * D_]);
}

// ---------------------------------------------------------------------------
// Fused KNN-memory + causal flash attention. One wave per 16-query tile.
// grid = (S/16, H, B), block = 32.
// Scores computed transposed (St = K @ Q^T) so softmax stats are lane-local
// and exp'd fragments are directly the A-fragment of P for the PV WMMA.
// ---------------------------------------------------------------------------
__global__ __launch_bounds__(32)
void attn_kernel(const float* __restrict__ qkv, const float* __restrict__ memk,
                 const float* __restrict__ memv, const int* __restrict__ idxb,
                 const __bf16* __restrict__ qh, const __bf16* __restrict__ kh,
                 const __bf16* __restrict__ vt, const float* __restrict__ gval,
                 float* __restrict__ ctx) {
  const int i0   = blockIdx.x * 16;
  const int h    = blockIdx.y;
  const int b    = blockIdx.z;
  const int lane = threadIdx.x;
  const int l15  = lane & 15;
  const int half = lane >> 4;

  __shared__ int   s_off[16][K_];
  __shared__ float s_p[16][K_];
  __shared__ float s_bc[16];

  // Gather row offsets for this tile's KNN indices
  for (int t = lane; t < 16 * K_; t += 32) {
    const int qq = t >> 5, j = t & 31;
    s_off[qq][j] = idxb[((long)(b * S_ + i0 + qq)) * K_ + j] * D_;
  }
  __syncthreads();

  // ---- memory branch scores (VALU): lane pair (q, q+16) splits 32 slots ----
  const float* qrow = qkv + (long)(b * S_ + i0 + l15) * (3 * D_) + h * DH_;
  const float* mkb  = memk + (long)b * M_ * D_ + h * DH_;
  float sj[16];
#pragma unroll 1
  for (int jj = 0; jj < 16; ++jj) {
    const int j = half * 16 + jj;
    const float* krow = mkb + s_off[l15][j];
    float s = 0.0f;
#pragma unroll 4
    for (int d = 0; d < DH_; ++d) s += qrow[d] * krow[d];
    sj[jj] = s * 0.125f;
  }
  float m = -3.0e38f;
#pragma unroll
  for (int jj = 0; jj < 16; ++jj) m = fmaxf(m, sj[jj]);
  m = fmaxf(m, __shfl_xor(m, 16));
  float l = 0.0f;
#pragma unroll
  for (int jj = 0; jj < 16; ++jj) {
    float p = __expf(sj[jj] - m);
    l += p;
    s_p[l15][half * 16 + jj] = p;
  }
  l += __shfl_xor(l, 16);
  __syncthreads();

  // ---- memory branch value accumulation, in WMMA O-layout ----
  v8f am0 = {}, am1 = {}, am2 = {}, am3 = {};
  const float* mvb = memv + (long)b * M_ * D_ + h * DH_ + l15;
#pragma unroll 1
  for (int r = 0; r < 8; ++r) {
    const int q = r + half * 8;
#pragma unroll 1
    for (int j = 0; j < K_; ++j) {
      const float p = s_p[q][j];
      const float* vr = mvb + s_off[q][j];
      am0[r] += p * vr[0];
      am1[r] += p * vr[16];
      am2[r] += p * vr[32];
      am3[r] += p * vr[48];
    }
  }

  // ---- causal flash loop over local keys, 32 keys (2 tiles) per step ----
  v8f al0 = {}, al1 = {}, al2 = {}, al3 = {};
  const __bf16* qhb = qh + (long)(b * H_ + h) * S_ * DH_;
  const __bf16* khb = kh + (long)(b * H_ + h) * S_ * DH_;
  const __bf16* vtb = vt + (long)(b * H_ + h) * DH_ * S_;

  const v16bf qb0 = packA(*(const v8bf*)(qhb + (long)(i0 + l15) * DH_ + half * 8), *(const v8bf*)(qhb + (long)(i0 + l15) * DH_ + 16 + half * 8));
  const v16bf qb1 = packA(*(const v8bf*)(qhb + (long)(i0 + l15) * DH_ + 32 + half * 8), *(const v8bf*)(qhb + (long)(i0 + l15) * DH_ + 48 + half * 8));

  const int npair = (i0 + 16 + 31) / 32;
  const int qg = i0 + l15;

  for (int kp = 0; kp < npair; ++kp) {
    const int k0 = kp * 32;
    v8f st0 = {}, st1 = {};
    {
      const __bf16* kr = khb + (long)(k0 + l15) * DH_ + half * 8;
      st0 = wmma_bf16(packA(*(const v8bf*)kr, *(const v8bf*)(kr + 16)), qb0, st0);
      st0 = wmma_bf16(packA(*(const v8bf*)(kr + 32), *(const v8bf*)(kr + 48)), qb1, st0);
    }
    {
      const __bf16* kr = khb + (long)(k0 + 16 + l15) * DH_ + half * 8;
      st1 = wmma_bf16(packA(*(const v8bf*)kr, *(const v8bf*)(kr + 16)), qb0, st1);
      st1 = wmma_bf16(packA(*(const v8bf*)(kr + 32), *(const v8bf*)(kr + 48)), qb1, st1);
    }
    // causal mask (St layout: lane = query, VGPR r = key within tile)
    if (k0 + 15 > i0) {
#pragma unroll
      for (int r = 0; r < 8; ++r) {
        if (k0 + r + half * 8 > qg) st0[r] = -3.0e38f;
      }
    }
    if (k0 + 31 > i0) {
#pragma unroll
      for (int r = 0; r < 8; ++r) {
        if (k0 + 16 + r + half * 8 > qg) st1[r] = -3.0e38f;
      }
    }
    // online softmax stats (per query = per lane pair)
    float tmax = -3.0e38f;
#pragma unroll
    for (int r = 0; r < 8; ++r) { tmax = fmaxf(tmax, st0[r]); tmax = fmaxf(tmax, st1[r]); }
    tmax = fmaxf(tmax, __shfl_xor(tmax, 16));
    const float mnew = fmaxf(m, tmax);
    const float scale = __expf(m - mnew);
    m = mnew;
    float lsum = 0.0f;
#pragma unroll
    for (int r = 0; r < 8; ++r) {
      st0[r] = __expf(st0[r] - m); lsum += st0[r];
      st1[r] = __expf(st1[r] - m); lsum += st1[r];
    }
    lsum += __shfl_xor(lsum, 16);
    l = l * scale + lsum;

    // broadcast per-query rescale factor to O-layout lanes
    if (lane < 16) s_bc[lane] = scale;
    __syncthreads();
#pragma unroll
    for (int r = 0; r < 8; ++r) {
      const float sc = s_bc[r + half * 8];
      al0[r] *= sc; al1[r] *= sc; al2[r] *= sc; al3[r] *= sc;
      am0[r] *= sc; am1[r] *= sc; am2[r] *= sc; am3[r] *= sc;
    }
    __syncthreads();

    // P A-fragment: lane-local bf16 convert of exp'd St fragments
    v16bf ap;
#pragma unroll
    for (int e = 0; e < 8; ++e) { ap[e] = (__bf16)st0[e]; ap[e + 8] = (__bf16)st1[e]; }

    const __bf16* vrow = vtb + (long)l15 * S_ + k0 + half * 8;
    al0 = wmma_bf16(ap, packA(*(const v8bf*)(vrow), *(const v8bf*)(vrow + 16)), al0);
    al1 = wmma_bf16(ap, packA(*(const v8bf*)(vrow + 16 * S_), *(const v8bf*)(vrow + 16 * S_ + 16)), al1);
    al2 = wmma_bf16(ap, packA(*(const v8bf*)(vrow + 32 * S_), *(const v8bf*)(vrow + 32 * S_ + 16)), al2);
    al3 = wmma_bf16(ap, packA(*(const v8bf*)(vrow + 48 * S_), *(const v8bf*)(vrow + 48 * S_ + 16)), al3);
  }

  // final combine: out = ((1-g)*local + g*mem) / l, stored as (B,S,H*DH)
  __syncthreads();
  if (lane < 16) s_bc[lane] = l;
  __syncthreads();
  const float g = gval[h];
  float* crow = ctx + (long)b * S_ * D_ + h * DH_;
  v8f AL[4] = {al0, al1, al2, al3};
  v8f AM[4] = {am0, am1, am2, am3};
#pragma unroll
  for (int r = 0; r < 8; ++r) {
    const int q = r + half * 8;
    const float linv = 1.0f / s_bc[q];
#pragma unroll
    for (int t = 0; t < 4; ++t) {
      const float o = (AL[t][r] * (1.0f - g) + AM[t][r] * g) * linv;
      crow[(long)(i0 + q) * D_ + t * 16 + l15] = o;
    }
  }
}

// ---------------------------------------------------------------------------
// Utility conversions
// ---------------------------------------------------------------------------
__global__ __launch_bounds__(256)
void f2bf_kernel(const float* __restrict__ in, __bf16* __restrict__ out, long n) {
  const long i = (long)blockIdx.x * 256 + threadIdx.x;
  if (i < n) out[i] = (__bf16)in[i];
}

// W (Kd x N) row-major fp32  ->  Wt (N x Kd) row-major bf16
__global__ __launch_bounds__(256)
void transpose_bf_kernel(const float* __restrict__ in, __bf16* __restrict__ out,
                         int N, int Kd) {
  const long o = (long)blockIdx.x * 256 + threadIdx.x;
  if (o >= (long)N * Kd) return;
  const int n = (int)(o / Kd);
  const int k = (int)(o % Kd);
  out[o] = (__bf16)in[(long)k * N + n];
}

// ---------------------------------------------------------------------------
extern "C" void kernel_launch(void* const* d_in, const int* in_sizes, int n_in,
                              void* d_out, int out_size, void* d_ws, size_t ws_size,
                              hipStream_t stream) {
  const float* x      = (const float*)d_in[0];
  const float* mem_k  = (const float*)d_in[1];
  const float* mem_v  = (const float*)d_in[2];
  const float* g_val  = (const float*)d_in[3];
  const float* ln1_g  = (const float*)d_in[4];
  const float* ln1_b  = (const float*)d_in[5];
  const float* W_attn = (const float*)d_in[6];
  const float* b_attn = (const float*)d_in[7];
  const float* W_proj = (const float*)d_in[8];
  const float* b_proj = (const float*)d_in[9];
  const float* ln2_g  = (const float*)d_in[10];
  const float* ln2_b  = (const float*)d_in[11];
  const float* W_fc   = (const float*)d_in[12];
  const float* b_fc   = (const float*)d_in[13];
  const float* W_out  = (const float*)d_in[14];
  const float* b_out  = (const float*)d_in[15];
  float* out = (float*)d_out;

  // Workspace bump allocator (256B aligned)
  char* wp = (char*)d_ws;
  auto alloc = [&](size_t bytes) -> void* {
    void* p = (void*)wp;
    wp += (bytes + 255) & ~(size_t)255;
    return p;
  };
  const long BS = (long)B_ * S_;                     // 2048 rows
  __bf16* wtA    = (__bf16*)alloc((size_t)3 * D_ * D_ * 2);        // (3D, D)
  __bf16* wtP    = (__bf16*)alloc((size_t)D_ * D_ * 2);            // (D, D)
  __bf16* wtF    = (__bf16*)alloc((size_t)4 * D_ * D_ * 2);        // (4D, D)
  __bf16* wtO    = (__bf16*)alloc((size_t)D_ * 4 * D_ * 2);        // (D, 4D)
  __bf16* memkbf = (__bf16*)alloc((size_t)B_ * M_ * D_ * 2);
  __bf16* h1bf   = (__bf16*)alloc((size_t)BS * D_ * 2);
  float*  qkv    = (float*) alloc((size_t)BS * 3 * D_ * 4);
  __bf16* sqbf   = (__bf16*)alloc((size_t)BS * D_ * 2);
  float*  sims   = (float*) alloc((size_t)BS * M_ * 4);
  int*    idx    = (int*)   alloc((size_t)BS * K_ * 4);
  __bf16* qhbf   = (__bf16*)alloc((size_t)BS * D_ * 2);
  __bf16* khbf   = (__bf16*)alloc((size_t)BS * D_ * 2);
  __bf16* vtbf   = (__bf16*)alloc((size_t)BS * D_ * 2);
  float*  ctx    = (float*) alloc((size_t)BS * D_ * 4);
  __bf16* ctxbf  = (__bf16*)alloc((size_t)BS * D_ * 2);
  float*  h2     = (float*) alloc((size_t)BS * D_ * 4);
  __bf16* ln2bf  = (__bf16*)alloc((size_t)BS * D_ * 2);
  __bf16* gelubf = (__bf16*)alloc((size_t)BS * 4 * D_ * 2);

  // 1) weight transpose+convert to bf16
  {
    long n;
    n = (long)3 * D_ * D_;
    transpose_bf_kernel<<<(n + 255) / 256, 256, 0, stream>>>(W_attn, wtA, 3 * D_, D_);
    n = (long)D_ * D_;
    transpose_bf_kernel<<<(n + 255) / 256, 256, 0, stream>>>(W_proj, wtP, D_, D_);
    n = (long)4 * D_ * D_;
    transpose_bf_kernel<<<(n + 255) / 256, 256, 0, stream>>>(W_fc, wtF, 4 * D_, D_);
    n = (long)D_ * 4 * D_;
    transpose_bf_kernel<<<(n + 255) / 256, 256, 0, stream>>>(W_out, wtO, D_, 4 * D_);
  }
  // 2) memory key DB -> bf16
  {
    long n = (long)B_ * M_ * D_;
    f2bf_kernel<<<(n + 255) / 256, 256, 0, stream>>>(mem_k, memkbf, n);
  }
  // 3) LN1
  ln_kernel<<<BS, 256, 0, stream>>>(x, ln1_g, ln1_b, h1bf);
  // 4) QKV GEMM: (2048 x 1024) @ (1024 x 3072) + b_attn
  gemm_nt<0><<<dim3(3 * D_ / 64, BS / 16, 1), 32, 0, stream>>>(
      h1bf, wtA, b_attn, nullptr, qkv, nullptr, (int)BS, 3 * D_, D_, 0, 0, 0);
  // 5) normalized query rows -> bf16
  rownorm_kernel<<<BS, 256, 0, stream>>>(qkv, sqbf);
  // 6) sims = sq @ mem_k^T, batched over B
  gemm_nt<0><<<dim3(M_ / 64, S_ / 16, B_), 32, 0, stream>>>(
      sqbf, memkbf, nullptr, nullptr, sims, nullptr, S_, M_, D_,
      (long)S_ * D_, (long)M_ * D_, (long)S_ * M_);
  // 7) top-K indices
  topk_kernel<<<BS, 256, 0, stream>>>(sims, idx);
  // 8) repack q/k/v per head (q scaled by 1/sqrt(DH), v transposed)
  {
    long n = (long)B_ * H_ * S_ * DH_;
    repack_kernel<<<(n + 255) / 256, 256, 0, stream>>>(qkv, qhbf, khbf, vtbf);
  }
  // 9) fused KNN-memory + causal flash attention
  attn_kernel<<<dim3(S_ / 16, H_, B_), 32, 0, stream>>>(
      qkv, mem_k, mem_v, idx, qhbf, khbf, vtbf, g_val, ctx);
  // 10) attention context -> bf16
  {
    long n = BS * D_;
    f2bf_kernel<<<(n + 255) / 256, 256, 0, stream>>>(ctx, ctxbf, n);
  }
  // 11) proj GEMM + bias + residual(x) -> h2
  gemm_nt<0><<<dim3(D_ / 64, BS / 16, 1), 32, 0, stream>>>(
      ctxbf, wtP, b_proj, x, h2, nullptr, (int)BS, D_, D_, 0, 0, 0);
  // 12) LN2
  ln_kernel<<<BS, 256, 0, stream>>>(h2, ln2_g, ln2_b, ln2bf);
  // 13) fc GEMM + bias + gelu -> bf16
  gemm_nt<1><<<dim3(4 * D_ / 64, BS / 16, 1), 32, 0, stream>>>(
      ln2bf, wtF, b_fc, nullptr, nullptr, gelubf, (int)BS, 4 * D_, D_, 0, 0, 0);
  // 14) out GEMM + bias + residual(h2) -> d_out
  gemm_nt<0><<<dim3(D_ / 64, BS / 16, 1), 32, 0, stream>>>(
      gelubf, wtO, b_out, h2, out, nullptr, (int)BS, D_, 4 * D_, 0, 0, 0);
}
